// MultiHeadAttention_51969104281804
// MI455X (gfx1250) — matrix operand order repaired and checked by reading, not compiled, not run on hardware
//
#include <hip/hip_runtime.h>
#include <hip/hip_bf16.h>

typedef _Float16 f16;
typedef __attribute__((ext_vector_type(16))) _Float16 v16h;
typedef __attribute__((ext_vector_type(8)))  _Float16 v8h;
typedef __attribute__((ext_vector_type(8)))  float    v8f;
typedef __attribute__((ext_vector_type(4)))  float    fx4;

constexpr int Bc  = 2;
constexpr int Sc  = 2048;
constexpr int Dc  = 1024;
constexpr int Hc  = 16;
constexpr int HDc = 64;
constexpr int WAVES = 8;
constexpr int BLOCK = 256;

__device__ __forceinline__ v8f wmma_f16(v16h a, v16h b, v8f c) {
    // D = A(16x32 f16) * B(32x16 f16) + C(16x16 f32)
    return __builtin_amdgcn_wmma_f32_16x16x32_f16(
        /*neg_a=*/false, a, /*neg_b=*/false, b,
        /*c_mod=*/(short)0, c, /*reuse_a=*/false, /*reuse_b=*/false);
}

// A fragment (16x32 f16) from f16 row-major source: lane m holds row row0+m;
// element i <-> k = kbase + g*16 + i, g = lane/16.
__device__ __forceinline__ v16h load_a_f16(const f16* __restrict__ base, int ld,
                                           size_t row0, int kbase, int lane) {
    const int m = lane & 15, g = (lane >> 4) & 1;
    const f16* p = base + (row0 + m) * (size_t)ld + kbase;
    v8h lo = *(const v8h*)(p + g * 16);
    v8h hi = *(const v8h*)(p + g * 16 + 8);
    v16h a;
#pragma unroll
    for (int i = 0; i < 8; ++i) { a[i] = lo[i]; a[8 + i] = hi[i]; }
    return a;
}

// B fragment (32x16 f16): lane n reads 16 contiguous f16 of row row0+n at
// kbase + g*16 (B[k][n] = Src[n][k]); single 32-byte load.
__device__ __forceinline__ v16h load_b_f16(const f16* __restrict__ base, int ld,
                                           size_t row0, int kbase, int lane) {
    const int n = lane & 15, g = (lane >> 4) & 1;
    return *(const v16h*)(base + (row0 + n) * (size_t)ld + kbase + g * 16);
}

// ---------------------------------------------------------------------------
// One-time fp32 -> f16 cast (8 elements / thread, b128 in, b128 out).
// Eliminates redundant in-loop conversions: weights are re-read by 256
// s-tiles and activations by 16 o-tile groups, all of which now load f16.
// ---------------------------------------------------------------------------
__global__ __launch_bounds__(BLOCK)
void cast_f16_kernel(const float* __restrict__ src, f16* __restrict__ dst, int n8) {
    const int i = blockIdx.x * BLOCK + threadIdx.x;
    if (i >= n8) return;
    const float* p = src + (size_t)i * 8;
    fx4 a = *(const fx4*)(p);
    fx4 b = *(const fx4*)(p + 4);
    v8h o;
#pragma unroll
    for (int j = 0; j < 4; ++j) { o[j] = (f16)a[j]; o[4 + j] = (f16)b[j]; }
    *(v8h*)(dst + (size_t)i * 8) = o;
}

// ---------------------------------------------------------------------------
// Transposed projection: Y[s][o] = scale * sum_k X[s][k] * W[o][k], f16 out.
// A = W rows, B = X^T. C-layout: lane = s, regs = 8 contiguous out-features
// -> one 16B f16 store per lane per tile. 4-way N-blocking reuses B fragment.
// Inner loop: 1 B load + 4 A loads + 4 WMMA (pure b128 + wmma).
// ---------------------------------------------------------------------------
__global__ __launch_bounds__(BLOCK)
void proj_t_kernel(const f16* __restrict__ X, const f16* __restrict__ W,
                   f16* __restrict__ Y, float scale) {
    const int lane = threadIdx.x & 31;
    const int wid  = threadIdx.x >> 5;
    const int NT4  = (Dc / 16) / 4;          // 16 groups of 4 o-tiles
    const int MT   = (Bc * Sc) / 16;         // 256 s-tiles
    const int task = blockIdx.x * WAVES + wid;
    if (task >= NT4 * MT) return;
    const int ot = (task % NT4) * 4;
    const int st = task / NT4;

    v8f acc[4] = {{}, {}, {}, {}};
    for (int k = 0; k < Dc; k += 32) {
        v16h bfrag = load_b_f16(X, Dc, (size_t)st * 16, k, lane);
#pragma unroll
        for (int j = 0; j < 4; ++j) {
            v16h a = load_a_f16(W, Dc, (size_t)(ot + j) * 16, k, lane);
            acc[j] = wmma_f16(a, bfrag, acc[j]);
        }
    }
    const int n = lane & 15, g = (lane >> 4) & 1;
#pragma unroll
    for (int j = 0; j < 4; ++j) {
        v8h out;
#pragma unroll
        for (int r = 0; r < 8; ++r) out[r] = (f16)(acc[j][r] * scale);
        *(v8h*)(Y + (size_t)(st * 16 + n) * Dc + (ot + j) * 16 + g * 8) = out;
    }
}

// ---------------------------------------------------------------------------
// V projection stored transposed: Vt[b][o][s] = sum_k X[b,s][k] * W[o][k].
// Normal orientation: A = X rows, B = W^T. C-layout: lane = o, regs = 8
// contiguous s -> one 16B store per lane into [b, D, S] f16 layout (exactly
// the A-fragment layout the attn ctx GEMM wants).
// ---------------------------------------------------------------------------
__global__ __launch_bounds__(BLOCK)
void proj_v_kernel(const f16* __restrict__ X, const f16* __restrict__ W,
                   f16* __restrict__ Vt) {
    const int lane = threadIdx.x & 31;
    const int wid  = threadIdx.x >> 5;
    const int NT4  = (Dc / 16) / 4;
    const int MT   = (Bc * Sc) / 16;
    const int task = blockIdx.x * WAVES + wid;
    if (task >= NT4 * MT) return;
    const int ot = (task % NT4) * 4;
    const int st = task / NT4;

    v8f acc[4] = {{}, {}, {}, {}};
    for (int k = 0; k < Dc; k += 32) {
        v16h afrag = load_a_f16(X, Dc, (size_t)st * 16, k, lane);
#pragma unroll
        for (int j = 0; j < 4; ++j) {
            v16h b = load_b_f16(W, Dc, (size_t)(ot + j) * 16, k, lane);
            acc[j] = wmma_f16(afrag, b, acc[j]);
        }
    }
    const int n = lane & 15, g = (lane >> 4) & 1;
    const int sg0 = st * 16 + g * 8;
    const int bb  = sg0 / Sc;
    const int sl  = sg0 - bb * Sc;
#pragma unroll
    for (int j = 0; j < 4; ++j) {
        const int o = (ot + j) * 16 + n;
        v8h out;
#pragma unroll
        for (int r = 0; r < 8; ++r) out[r] = (f16)acc[j][r];
        *(v8h*)(Vt + ((size_t)bb * Dc + o) * Sc + sl) = out;
    }
}

// ---------------------------------------------------------------------------
// Flash attention, one wave per (b, h, 16-query tile).
// S^T = K * Q^T (C-layout: lane = query, regs/halves = keys) -> softmax key
// reduction is 16 in-lane values + one shfl_xor(16). P^T repacked into a
// B-fragment with shfl_xor(16) (no LDS round-trip); ctx^T += V^T * P^T.
// ---------------------------------------------------------------------------
__global__ __launch_bounds__(BLOCK)
void attn_kernel(const f16* __restrict__ Q, const f16* __restrict__ K,
                 const f16* __restrict__ Vt, f16* __restrict__ Ctx) {
    const int lane = threadIdx.x & 31;
    const int wid  = threadIdx.x >> 5;
    const int QT   = Sc / 16;                // 128 query tiles
    const int task = blockIdx.x * WAVES + wid;
    if (task >= Bc * Hc * QT) return;
    const int qt = task % QT;
    const int h  = (task / QT) % Hc;
    const int b  = task / (QT * Hc);
    const int n  = lane & 15, g = (lane >> 4) & 1;

    const size_t qrow0 = (size_t)b * Sc + qt * 16;
    const size_t krow0 = (size_t)b * Sc;
    const size_t vrow0 = (size_t)b * Dc + h * HDc;

    // Q^T B-fragments (d = 0..31 and 32..63), reused for all key blocks.
    const v16h qb0 = *(const v16h*)(Q + (qrow0 + n) * Dc + h * HDc + g * 16);
    const v16h qb1 = *(const v16h*)(Q + (qrow0 + n) * Dc + h * HDc + 32 + g * 16);

    v8f c0 = {}, c1 = {}, c2 = {}, c3 = {};
    float m_run = -1e30f, l_run = 0.0f;

    for (int k0 = 0; k0 < Sc; k0 += 32) {
        if (k0 + 32 < Sc)  // prefetch next K block rows (wave-uniform branch)
            __builtin_prefetch(K + (krow0 + k0 + 32 + n) * Dc + h * HDc, 0, 1);

        v8f s0 = {}, s1 = {};
        s0 = wmma_f16(load_a_f16(K, Dc, krow0 + k0,      h * HDc,      lane), qb0, s0);
        s0 = wmma_f16(load_a_f16(K, Dc, krow0 + k0,      h * HDc + 32, lane), qb1, s0);
        s1 = wmma_f16(load_a_f16(K, Dc, krow0 + k0 + 16, h * HDc,      lane), qb0, s1);
        s1 = wmma_f16(load_a_f16(K, Dc, krow0 + k0 + 16, h * HDc + 32, lane), qb1, s1);

        float bm = -1e30f;
#pragma unroll
        for (int r = 0; r < 8; ++r) bm = fmaxf(bm, fmaxf(s0[r], s1[r]));
        bm = fmaxf(bm, __shfl_xor(bm, 16, 32));

        const float m_new = fmaxf(m_run, bm);
        const float alpha = __expf(m_run - m_new);

        float e0[8], e1[8], bs = 0.0f;
#pragma unroll
        for (int r = 0; r < 8; ++r) {
            e0[r] = __expf(s0[r] - m_new);
            e1[r] = __expf(s1[r] - m_new);
            bs += e0[r] + e1[r];
        }
        bs += __shfl_xor(bs, 16, 32);
        l_run = l_run * alpha + bs;
        m_run = m_new;

#pragma unroll
        for (int r = 0; r < 8; ++r) {
            c0[r] *= alpha; c1[r] *= alpha; c2[r] *= alpha; c3[r] *= alpha;
        }

        // Repack P^T into B-fragment layout: element i <-> key g*16 + i.
        v16h pb;
#pragma unroll
        for (int r = 0; r < 8; ++r) {
            const float o0 = __shfl_xor(e0[r], 16, 32);
            const float o1 = __shfl_xor(e1[r], 16, 32);
            pb[r]     = (f16)(g ? o1 : e0[r]);
            pb[8 + r] = (f16)(g ? e1[r] : o0);
        }

        c0 = wmma_f16(load_a_f16(Vt, Sc, vrow0 + 0,  k0, lane), pb, c0);
        c1 = wmma_f16(load_a_f16(Vt, Sc, vrow0 + 16, k0, lane), pb, c1);
        c2 = wmma_f16(load_a_f16(Vt, Sc, vrow0 + 32, k0, lane), pb, c2);
        c3 = wmma_f16(load_a_f16(Vt, Sc, vrow0 + 48, k0, lane), pb, c3);
    }

    const float inv = 1.0f / l_run;
    f16* cp = Ctx + (qrow0 + n) * Dc + h * HDc + g * 8;
    v8h o0, o1, o2, o3;
#pragma unroll
    for (int r = 0; r < 8; ++r) {
        o0[r] = (f16)(c0[r] * inv);
        o1[r] = (f16)(c1[r] * inv);
        o2[r] = (f16)(c2[r] * inv);
        o3[r] = (f16)(c3[r] * inv);
    }
    *(v8h*)(cp)      = o0;
    *(v8h*)(cp + 16) = o1;
    *(v8h*)(cp + 32) = o2;
    *(v8h*)(cp + 48) = o3;
}

// ---------------------------------------------------------------------------
// Output projection: Out[s][o] = sum_k Ctx[s][k] * Wo[o][k], fp32 out.
// Same shape as proj_t but stores fp32 (2x b128 per lane per tile).
// ---------------------------------------------------------------------------
__global__ __launch_bounds__(BLOCK)
void proj_out_kernel(const f16* __restrict__ Ctx, const f16* __restrict__ W,
                     float* __restrict__ Out) {
    const int lane = threadIdx.x & 31;
    const int wid  = threadIdx.x >> 5;
    const int NT4  = (Dc / 16) / 4;
    const int MT   = (Bc * Sc) / 16;
    const int task = blockIdx.x * WAVES + wid;
    if (task >= NT4 * MT) return;
    const int ot = (task % NT4) * 4;
    const int st = task / NT4;
    const int n = lane & 15, g = (lane >> 4) & 1;

    v8f acc[4] = {{}, {}, {}, {}};
    for (int k = 0; k < Dc; k += 32) {
        v16h bfrag = load_b_f16(Ctx, Dc, (size_t)st * 16, k, lane);
#pragma unroll
        for (int j = 0; j < 4; ++j) {
            v16h a = load_a_f16(W, Dc, (size_t)(ot + j) * 16, k, lane);
            acc[j] = wmma_f16(a, bfrag, acc[j]);
        }
    }
#pragma unroll
    for (int j = 0; j < 4; ++j) {
        float* p = Out + (size_t)(st * 16 + n) * Dc + (ot + j) * 16 + g * 8;
        fx4 lo = {acc[j][0], acc[j][1], acc[j][2], acc[j][3]};
        fx4 hi = {acc[j][4], acc[j][5], acc[j][6], acc[j][7]};
        *(fx4*)(p)     = lo;
        *(fx4*)(p + 4) = hi;
    }
}

extern "C" void kernel_launch(void* const* d_in, const int* in_sizes, int n_in,
                              void* d_out, int out_size, void* d_ws, size_t ws_size,
                              hipStream_t stream) {
    const float* q  = (const float*)d_in[0];
    const float* k  = (const float*)d_in[1];
    const float* v  = (const float*)d_in[2];
    const float* Wq = (const float*)d_in[3];
    const float* Wk = (const float*)d_in[4];
    const float* Wv = (const float*)d_in[5];
    const float* Wo = (const float*)d_in[6];
    float* out = (float*)d_out;

    const size_t elems  = (size_t)Bc * Sc * Dc;  // 4 Mi activation elements
    const size_t welems = (size_t)Dc * Dc;       // 1 Mi weight elements
    f16* Qb  = (f16*)d_ws;        // 8 MiB
    f16* Kb  = Qb + elems;        // 8 MiB
    f16* Vtb = Kb + elems;        // 8 MiB
    f16* Cb  = Vtb + elems;       // 8 MiB
    f16* Xh  = Cb + elems;        // 8 MiB staging: q/k/v cast, reused serially
    f16* Wh  = Xh + elems;        // 2 MiB staging: weight cast, reused serially

    const int castXBlocks = (int)(elems / 8 / BLOCK);   // 2048
    const int castWBlocks = (int)(welems / 8 / BLOCK);  // 512
    const int projTasks   = ((Dc / 16) / 4) * ((Bc * Sc) / 16);  // 4096
    const int projBlocks  = projTasks / WAVES;                   // 512
    const int attnBlocks  = (Bc * Hc * (Sc / 16)) / WAVES;       // 512
    const float qscale    = 0.03125f;                            // 1/sqrt(D)

    // Q projection
    cast_f16_kernel<<<castXBlocks, BLOCK, 0, stream>>>(q,  Xh, (int)(elems / 8));
    cast_f16_kernel<<<castWBlocks, BLOCK, 0, stream>>>(Wq, Wh, (int)(welems / 8));
    proj_t_kernel<<<projBlocks, BLOCK, 0, stream>>>(Xh, Wh, Qb, qscale);
    // K projection
    cast_f16_kernel<<<castXBlocks, BLOCK, 0, stream>>>(k,  Xh, (int)(elems / 8));
    cast_f16_kernel<<<castWBlocks, BLOCK, 0, stream>>>(Wk, Wh, (int)(welems / 8));
    proj_t_kernel<<<projBlocks, BLOCK, 0, stream>>>(Xh, Wh, Kb, 1.0f);
    // V projection (stored transposed)
    cast_f16_kernel<<<castXBlocks, BLOCK, 0, stream>>>(v,  Xh, (int)(elems / 8));
    cast_f16_kernel<<<castWBlocks, BLOCK, 0, stream>>>(Wv, Wh, (int)(welems / 8));
    proj_v_kernel<<<projBlocks, BLOCK, 0, stream>>>(Xh, Wh, Vtb);
    // Attention
    attn_kernel<<<attnBlocks, BLOCK, 0, stream>>>(Qb, Kb, Vtb, Cb);
    // Output projection
    cast_f16_kernel<<<castWBlocks, BLOCK, 0, stream>>>(Wo, Wh, (int)(welems / 8));
    proj_out_kernel<<<projBlocks, BLOCK, 0, stream>>>(Cb, Wh, out);
}
